// GRUDecoder_38079180046975
// MI455X (gfx1250) — matrix operand order repaired and checked by reading, not compiled, not run on hardware
//
#include <hip/hip_runtime.h>
#include <hip/hip_bf16.h>

#define HIDDEN 512
#define ENC    256
#define VOCAB  42
#define BATCH  4096
#define STEPS  128
#define BT     16      // batch rows per workgroup
#define NTHR   256     // 8 wave32 waves

typedef __attribute__((ext_vector_type(16))) __bf16 v16bf;
typedef __attribute__((ext_vector_type(8)))  __bf16 v8bf;
typedef __attribute__((ext_vector_type(8)))  float  v8f;

// ---------------------------------------------------------------------------
// Pack fp32 row-major W[N][K] into a WMMA bf16 B-fragment layout (contiguous k order, matching load_a):
// element e of lane L of fragment (nTile,kStep) = W[nTile*16+(L&15)][kStep*32+(L>>4)*16+e]
// Out-of-range N/K padded with zero (VOCAB 42 -> 64/48 padding).
// ---------------------------------------------------------------------------
__global__ void pack_b_kernel(const float* __restrict__ src, __bf16* __restrict__ dst,
                              int N, int K, int K32, int total) {
  int idx = blockIdx.x * blockDim.x + threadIdx.x;
  if (idx >= total) return;
  int e     = idx & 15;
  int lane  = (idx >> 4) & 31;
  int rest  = idx >> 9;
  int kStep = rest % K32;
  int nTile = rest / K32;
  int n = nTile * 16 + (lane & 15);
  int k = kStep * 32 + ((lane >> 4) << 4) + e;
  float v = (n < N && k < K) ? src[(size_t)n * K + k] : 0.0f;
  dst[idx] = (__bf16)v;
}

// A-fragment (16x32 bf16) load from LDS, contiguous order matching packed B:
// lane<16: elems 0..15 -> K 0..15 ; lane>=16: elems 0..15 -> K 16..31
__device__ __forceinline__ v16bf load_a(const __bf16* base, int ld, int lane, int kStep) {
  const __bf16* p = base + (lane & 15) * ld + kStep * 32 + ((lane >> 4) << 4);
  v8bf lo = *(const v8bf*)p;
  v8bf hi = *(const v8bf*)(p + 8);
  return __builtin_shufflevector(lo, hi, 0,1,2,3,4,5,6,7,8,9,10,11,12,13,14,15);
}

__device__ __forceinline__ v16bf load_b(const __bf16* __restrict__ pack,
                                        int nTile, int K32, int k, int lane) {
  return *(const v16bf*)(pack + (((size_t)nTile * K32 + k) * 32 + lane) * 16);
}

__device__ __forceinline__ v8f wmma_bf16(v16bf a, v16bf b, v8f c) {
  return __builtin_amdgcn_wmma_f32_16x16x32_bf16(false, a, false, b, (short)0, c, false, false);
}

__device__ __forceinline__ float sigm(float x) { return 1.0f / (1.0f + __expf(-x)); }

// ---------------------------------------------------------------------------
// One GRU layer for this wave's 4 hidden tiles (ht = wave + 8t).
// k-outer: one A fragment feeds 12 WMMAs. r/z accumulators are shared between
// the input-side and hidden-side GEMMs (GRU needs only i_r+h_r, i_z+h_z);
// only the n gate keeps separate i_n / h_n.
// ---------------------------------------------------------------------------
__device__ __forceinline__ void gru_layer(
    const __bf16* __restrict__ aIn, int aLd, int K32i,   // input acts (x or h0)
    const __bf16* __restrict__ hB,                       // hidden acts bf16 LDS
    const __bf16* __restrict__ pIh, const __bf16* __restrict__ pHh,
    int wave, int lane,
    const float* brz, const float* bzz, const float* bin_, const float* bhn,
    float (&hreg)[4][8], float (&hnew)[4][8]) {
  v8f r[4]  = {{}, {}, {}, {}};
  v8f z[4]  = {{}, {}, {}, {}};
  v8f ni[4] = {{}, {}, {}, {}};
  v8f nh[4] = {{}, {}, {}, {}};

  for (int k = 0; k < K32i; ++k) {                 // input-side GEMM
    v16bf a = load_a(aIn, aLd, lane, k);
    #pragma unroll
    for (int t = 0; t < 4; ++t) {
      int ht = wave + t * 8;
      r[t]  = wmma_bf16(a, load_b(pIh, ht,      K32i, k, lane), r[t]);
      z[t]  = wmma_bf16(a, load_b(pIh, 32 + ht, K32i, k, lane), z[t]);
      ni[t] = wmma_bf16(a, load_b(pIh, 64 + ht, K32i, k, lane), ni[t]);
    }
  }
  for (int k = 0; k < HIDDEN / 32; ++k) {          // hidden-side GEMM (K=512)
    v16bf a = load_a(hB, HIDDEN, lane, k);
    #pragma unroll
    for (int t = 0; t < 4; ++t) {
      int ht = wave + t * 8;
      r[t]  = wmma_bf16(a, load_b(pHh, ht,      16, k, lane), r[t]);   // i_r+h_r
      z[t]  = wmma_bf16(a, load_b(pHh, 32 + ht, 16, k, lane), z[t]);   // i_z+h_z
      nh[t] = wmma_bf16(a, load_b(pHh, 64 + ht, 16, k, lane), nh[t]);
    }
  }
  #pragma unroll
  for (int t = 0; t < 4; ++t) {
    #pragma unroll
    for (int e = 0; e < 8; ++e) {
      float rg = sigm(r[t][e] + brz[t]);
      float zg = sigm(z[t][e] + bzz[t]);
      float ng = tanhf(ni[t][e] + bin_[t] + rg * (nh[t][e] + bhn[t]));
      hnew[t][e] = (1.0f - zg) * ng + zg * hreg[t][e];
    }
  }
}

// ---------------------------------------------------------------------------
// Main: 256 WGs x 16 batch rows, 8 waves each own 4 hidden tiles for 128 steps.
// fp32 hidden state lives in registers (static tile ownership); bf16 copies in
// LDS are the WMMA A-operands. Weights streamed from L2 in packed bf16.
// ---------------------------------------------------------------------------
__global__ void __launch_bounds__(NTHR)
gru_decoder_kernel(const float* __restrict__ latent,
                   const __bf16* __restrict__ pFc1, const float* __restrict__ fc1_b,
                   const __bf16* __restrict__ pFc2, const float* __restrict__ fc2_b,
                   const __bf16* __restrict__ pIh0, const __bf16* __restrict__ pHh0,
                   const float* __restrict__ bih0, const float* __restrict__ bhh0,
                   const __bf16* __restrict__ pIh1, const __bf16* __restrict__ pHh1,
                   const float* __restrict__ bih1, const float* __restrict__ bhh1,
                   float* __restrict__ out) {
  __shared__ __bf16 h0b[BT * HIDDEN];   // 16 KB
  __shared__ __bf16 h1b[BT * HIDDEN];   // 16 KB
  __shared__ __bf16 xb[BT * 64];        //  2 KB (VOCAB padded to 64)
  __shared__ __bf16 latb[BT * ENC];     //  8 KB

  const int tid    = threadIdx.x;
  const int wave   = tid >> 5;
  const int lane   = tid & 31;
  const int wgBase = blockIdx.x * BT;
  const int colLo  = lane & 15;
  const int rowHi  = (lane >> 4) << 3;   // 0 or 8

  // ---- init LDS ----
  for (int i = tid; i < BT * ENC; i += NTHR)
    latb[i] = (__bf16)latent[(size_t)(wgBase + i / ENC) * ENC + (i % ENC)];
  for (int i = tid; i < BT * 64; i += NTHR)
    xb[i] = (__bf16)(((i & 63) == (VOCAB - 1)) ? 1.0f : 0.0f);   // start token one-hot
  for (int i = tid; i < BT * HIDDEN; i += NTHR)
    h1b[i] = (__bf16)0.0f;
  __syncthreads();

  // ---- hoist step-invariant biases (r/z pre-summed: only i_g+h_g is used) ----
  float l0brz[4], l0bz[4], l0bin[4], l0bhn[4];
  float l1brz[4], l1bz[4], l1bin[4], l1bhn[4];
  #pragma unroll
  for (int t = 0; t < 4; ++t) {
    int col = (wave + t * 8) * 16 + colLo;
    l0brz[t] = bih0[col]              + bhh0[col];
    l0bz[t]  = bih0[HIDDEN + col]     + bhh0[HIDDEN + col];
    l0bin[t] = bih0[2 * HIDDEN + col];
    l0bhn[t] = bhh0[2 * HIDDEN + col];
    l1brz[t] = bih1[col]              + bhh1[col];
    l1bz[t]  = bih1[HIDDEN + col]     + bhh1[HIDDEN + col];
    l1bin[t] = bih1[2 * HIDDEN + col];
    l1bhn[t] = bhh1[2 * HIDDEN + col];
  }
  const int fcol = wave * 16 + colLo;
  const float fb = (wave < 3 && fcol < VOCAB) ? fc2_b[fcol] : 0.0f;

  float h0reg[4][8];
  float h1reg[4][8];

  // ---- h0 = fc1(latent) + b : k-outer, 4 tile accumulators ----
  {
    v8f acc[4] = {{}, {}, {}, {}};
    for (int k = 0; k < ENC / 32; ++k) {
      v16bf a = load_a(latb, ENC, lane, k);
      #pragma unroll
      for (int t = 0; t < 4; ++t)
        acc[t] = wmma_bf16(a, load_b(pFc1, wave + t * 8, ENC / 32, k, lane), acc[t]);
    }
    #pragma unroll
    for (int t = 0; t < 4; ++t) {
      int col = (wave + t * 8) * 16 + colLo;
      float b = fc1_b[col];
      #pragma unroll
      for (int r = 0; r < 8; ++r) {
        float v = acc[t][r] + b;
        h0reg[t][r] = v;
        h1reg[t][r] = 0.0f;
        h0b[(r + rowHi) * HIDDEN + col] = (__bf16)v;
      }
    }
  }
  __syncthreads();

  for (int step = 0; step < STEPS; ++step) {
    float hnew[4][8];
    // ================= GRU layer 0 (input = x, 42->64 padded) =================
    gru_layer(xb, 64, 2, h0b, pIh0, pHh0, wave, lane,
              l0brz, l0bz, l0bin, l0bhn, h0reg, hnew);
    __syncthreads();                       // all reads of old h0b done
    #pragma unroll
    for (int t = 0; t < 4; ++t) {
      int col = (wave + t * 8) * 16 + colLo;
      #pragma unroll
      for (int r = 0; r < 8; ++r) {
        h0reg[t][r] = hnew[t][r];
        h0b[(r + rowHi) * HIDDEN + col] = (__bf16)hnew[t][r];
      }
    }
    __syncthreads();                       // new h0b visible

    // ================= GRU layer 1 (input = h0) =================
    gru_layer(h0b, HIDDEN, HIDDEN / 32, h1b, pIh1, pHh1, wave, lane,
              l1brz, l1bz, l1bin, l1bhn, h1reg, hnew);
    __syncthreads();
    #pragma unroll
    for (int t = 0; t < 4; ++t) {
      int col = (wave + t * 8) * 16 + colLo;
      #pragma unroll
      for (int r = 0; r < 8; ++r) {
        h1reg[t][r] = hnew[t][r];
        h1b[(r + rowHi) * HIDDEN + col] = (__bf16)hnew[t][r];
      }
    }
    __syncthreads();                       // new h1b visible

    // ================= fc2 logits + feedback x =================
    if (wave < 3) {                        // 48 output cols (42 + 6 zero-pad)
      v8f acc = {};
      for (int k = 0; k < HIDDEN / 32; ++k) {
        v16bf a = load_a(h1b, HIDDEN, lane, k);
        acc = wmma_bf16(a, load_b(pFc2, wave, HIDDEN / 32, k, lane), acc);
      }
      #pragma unroll
      for (int r = 0; r < 8; ++r) {
        int row = r + rowHi;
        float v = acc[r] + fb;
        if (fcol < VOCAB) {
          out[(((size_t)(wgBase + row)) * STEPS + step) * VOCAB + fcol] = v;
          xb[row * 64 + fcol] = (__bf16)v;
        } else {
          xb[row * 64 + fcol] = (__bf16)0.0f;
        }
      }
    }
    __syncthreads();                       // x ready for next step's layer 0
  }
}

// ---------------------------------------------------------------------------
extern "C" void kernel_launch(void* const* d_in, const int* in_sizes, int n_in,
                              void* d_out, int out_size, void* d_ws, size_t ws_size,
                              hipStream_t stream) {
  const float* latent = (const float*)d_in[0];
  const float* fc1_w  = (const float*)d_in[1];
  const float* fc1_b  = (const float*)d_in[2];
  const float* fc2_w  = (const float*)d_in[3];
  const float* fc2_b  = (const float*)d_in[4];
  const float* w_ih0  = (const float*)d_in[5];
  const float* w_hh0  = (const float*)d_in[6];
  const float* b_ih0  = (const float*)d_in[7];
  const float* b_hh0  = (const float*)d_in[8];
  const float* w_ih1  = (const float*)d_in[9];
  const float* w_hh1  = (const float*)d_in[10];
  const float* b_ih1  = (const float*)d_in[11];
  const float* b_hh1  = (const float*)d_in[12];

  // workspace: packed bf16 weight fragments (~5.0 MB total)
  __bf16* ws   = (__bf16*)d_ws;
  __bf16* pFc1 = ws;                 // 32 tiles * 8 ksteps * 512  = 131072
  __bf16* pFc2 = pFc1 + 131072;      //  3 * 16 * 512              =  24576
  __bf16* pIh0 = pFc2 + 24576;       // 96 *  2 * 512              =  98304
  __bf16* pHh0 = pIh0 + 98304;       // 96 * 16 * 512              = 786432
  __bf16* pIh1 = pHh0 + 786432;
  __bf16* pHh1 = pIh1 + 786432;

  auto pack = [&](const float* src, __bf16* dst, int N, int K, int K32, int total) {
    pack_b_kernel<<<(total + 255) / 256, 256, 0, stream>>>(src, dst, N, K, K32, total);
  };
  pack(fc1_w, pFc1, 512, 256, 8, 131072);
  pack(fc2_w, pFc2, VOCAB, 512, 16, 24576);
  pack(w_ih0, pIh0, 3 * HIDDEN, VOCAB, 2, 98304);
  pack(w_hh0, pHh0, 3 * HIDDEN, 512, 16, 786432);
  pack(w_ih1, pIh1, 3 * HIDDEN, 512, 16, 786432);
  pack(w_hh1, pHh1, 3 * HIDDEN, 512, 16, 786432);

  gru_decoder_kernel<<<BATCH / BT, NTHR, 0, stream>>>(
      latent, pFc1, fc1_b, pFc2, fc2_b,
      pIh0, pHh0, b_ih0, b_hh0,
      pIh1, pHh1, b_ih1, b_hh1,
      (float*)d_out);
}
